// MultiHeadedAttention_46952582480478
// MI455X (gfx1250) — matrix operand order repaired and checked by reading, not compiled, not run on hardware
//
#include <hip/hip_runtime.h>
#include <type_traits>
#include <cstddef>

#define S_LEN  2048
#define DMODEL 1024
#define NHEAD  16
#define DKH    64
#define RELSTRIDE 2049

typedef _Float16 h16 __attribute__((ext_vector_type(16)));
typedef _Float16 h8  __attribute__((ext_vector_type(8)));
typedef _Float16 h4  __attribute__((ext_vector_type(4)));
typedef float    f8  __attribute__((ext_vector_type(8)));

static __device__ __forceinline__ h16 cat8(h8 lo, h8 hi) {
  return __builtin_shufflevector(lo, hi, 0,1,2,3,4,5,6,7,8,9,10,11,12,13,14,15);
}
// A-matrix fragment (16x32 f16): lane = row M, 16 contiguous K halfs starting at 2*c, c = (lane/16)*8
static __device__ __forceinline__ h16 fragA(const _Float16* row, int c_lo) {
  h8 lo = *(const h8*)(row + 2 * c_lo);
  h8 hi = *(const h8*)(row + 2 * c_lo + 8);
  return cat8(lo, hi);
}
// B-matrix fragment (32x16 f16): lane = col N, 16 contiguous K halfs starting at (lane/16)*16
static __device__ __forceinline__ h16 fragContig(const _Float16* p) {
  h8 lo = *(const h8*)(p);
  h8 hi = *(const h8*)(p + 8);
  return cat8(lo, hi);
}
static __device__ __forceinline__ f8 wmma_f16(h16 a, h16 b, f8 c) {
  return __builtin_amdgcn_wmma_f32_16x16x32_f16(false, a, false, b, (short)0, c, false, false);
}

// CDNA5 async global->LDS copy, 16B per lane, tracked by ASYNCcnt (ISA 15.18.3 op 98)
static __device__ __forceinline__ void async_cp16(_Float16* lds_dst, const _Float16* gsrc) {
  _Float16 __attribute__((address_space(3)))* l3 =
      (_Float16 __attribute__((address_space(3)))*)lds_dst;
  const unsigned loff = (unsigned)(unsigned long long)l3;
  asm volatile("global_load_async_to_lds_b128 %0, %1, off"
               :: "v"(loff), "v"((unsigned long long)gsrc)
               : "memory");
}

// ---------------------------------------------------------------------------
// Tiled GEMM: out(M=2048 x N=1024) = X(2048x1024) @ W^T(1024x1024) + bias
// 128x128 tile, K-tile 64, 8 waves (4x2, 32x64 each), register double-buffer:
// stage k+1 global loads fly while stage k's 16 WMMAs execute.
// MODE 0: -> q_h[h][s][dk] f16, scaled by 1/8       MODE 1: -> k_h[h][s][dk] f16
// MODE 2: -> vT[h][dk][s]  f16 (pre-transposed)     MODE 3: -> f32 out[s][o]
// ---------------------------------------------------------------------------
template <int MODE, typename TX>
__global__ __launch_bounds__(256) void gemm_wmma_kernel(
    const TX* __restrict__ X, const float* __restrict__ W,
    const float* __restrict__ bias, void* __restrict__ outp)
{
  constexpr int K = DMODEL;
  using AElem = typename std::conditional<std::is_same<TX, float>::value, float4, h4>::type;
  __shared__ __align__(16) _Float16 As[128][72];   // 64 + 8 pad: 144B rows, conflict-free b128
  __shared__ __align__(16) _Float16 Bs[128][72];

  const int t    = threadIdx.x;
  const int wave = t >> 5;
  const int lane = t & 31;
  const int lr   = lane & 15;
  const int hf   = lane >> 4;
  const int m0   = blockIdx.y * 128;
  const int n0   = blockIdx.x * 128;
  const int wm   = (wave >> 1) * 32;
  const int wn   = (wave & 1) * 64;

  f8 acc[2][4];
  #pragma unroll
  for (int mt = 0; mt < 2; ++mt)
    #pragma unroll
    for (int nt = 0; nt < 4; ++nt)
      #pragma unroll
      for (int r = 0; r < 8; ++r) acc[mt][nt][r] = 0.0f;

  AElem  ra[8];
  float4 rw[8];
  auto load_tile = [&](int k0) {
    #pragma unroll
    for (int p = 0; p < 8; ++p) {
      const int qi  = t + p * 256;
      const int row = qi >> 4;
      const int c4  = (qi & 15) << 2;
      ra[p] = *(const AElem*)(X + (size_t)(m0 + row) * K + k0 + c4);
      rw[p] = *(const float4*)(W + (size_t)(n0 + row) * K + k0 + c4);
    }
  };

  load_tile(0);
  for (int k0 = 0; k0 < K; k0 += 64) {
    __syncthreads();
    #pragma unroll
    for (int p = 0; p < 8; ++p) {               // regs -> LDS (convert f32 -> f16)
      const int qi  = t + p * 256;
      const int row = qi >> 4;
      const int c4  = (qi & 15) << 2;
      _Float16* da = &As[row][c4];
      if constexpr (std::is_same<TX, float>::value) {
        da[0]=(_Float16)ra[p].x; da[1]=(_Float16)ra[p].y;
        da[2]=(_Float16)ra[p].z; da[3]=(_Float16)ra[p].w;
      } else {
        *(h4*)da = ra[p];
      }
      _Float16* db = &Bs[row][c4];
      db[0]=(_Float16)rw[p].x; db[1]=(_Float16)rw[p].y;
      db[2]=(_Float16)rw[p].z; db[3]=(_Float16)rw[p].w;
    }
    __syncthreads();

    if (k0 + 64 < K) load_tile(k0 + 64);        // prefetch next stage into VGPRs

    #pragma unroll
    for (int kc = 0; kc < 2; ++kc) {            // two 32-wide K chunks per stage
      h16 af[2];
      #pragma unroll
      for (int mt = 0; mt < 2; ++mt)
        af[mt] = fragA(&As[wm + mt*16 + lr][kc * 32], hf * 8);
      #pragma unroll
      for (int nt = 0; nt < 4; ++nt) {
        h16 bf = fragContig(&Bs[wn + nt*16 + lr][kc * 32 + hf * 16]);
        #pragma unroll
        for (int mt = 0; mt < 2; ++mt)
          acc[mt][nt] = wmma_f16(af[mt], bf, acc[mt][nt]);
      }
    }
  }

  #pragma unroll
  for (int nt = 0; nt < 4; ++nt) {
    const int gn = n0 + wn + nt*16 + lr;
    const float bv = bias[gn];
    #pragma unroll
    for (int mt = 0; mt < 2; ++mt) {
      #pragma unroll
      for (int r = 0; r < 8; ++r) {
        const int gm = m0 + wm + mt*16 + r + hf*8;
        const float v = acc[mt][nt][r] + bv;
        if constexpr (MODE == 0) {
          const int h = gn >> 6, dk = gn & 63;
          ((_Float16*)outp)[((size_t)h * S_LEN + gm) * DKH + dk] = (_Float16)(v * 0.125f);
        } else if constexpr (MODE == 1) {
          const int h = gn >> 6, dk = gn & 63;
          ((_Float16*)outp)[((size_t)h * S_LEN + gm) * DKH + dk] = (_Float16)v;
        } else if constexpr (MODE == 2) {
          const int h = gn >> 6, dk = gn & 63;
          ((_Float16*)outp)[((size_t)h * DKH + dk) * S_LEN + gm] = (_Float16)v;
        } else {
          ((float*)outp)[(size_t)gm * DMODEL + gn] = v;
        }
      }
    }
  }
}

// ---------------------------------------------------------------------------
// rel bias: rb[h,j] = (q_scaled[h,j,:] . rel_embed[h,j,:])   (scale already in q)
// ---------------------------------------------------------------------------
__global__ __launch_bounds__(256) void rel_bias_kernel(
    const _Float16* __restrict__ qh, const float* __restrict__ rel,
    float* __restrict__ rb)
{
  const int id = blockIdx.x * 256 + threadIdx.x;      // H*S threads
  const int h = id >> 11;
  const int j = id & (S_LEN - 1);
  const _Float16* q = qh + ((size_t)h * S_LEN + j) * DKH;
  const float* re = rel + ((size_t)h * RELSTRIDE + j) * DKH;
  float acc = 0.f;
  #pragma unroll 8
  for (int d = 0; d < DKH; ++d) acc += (float)q[d] * re[d];
  rb[id] = acc;
}

// ---------------------------------------------------------------------------
// Flash attention: 4 waves/block = 64 consecutive queries of one head.
// 64-key K/V tiles async-copied (ASYNCcnt) into double-buffered LDS, shared by
// the 4 waves; online softmax with the row-sum computed as a WMMA accumulator
// against a ones-matrix (no sum shuffle tree).
// ---------------------------------------------------------------------------
__global__ __launch_bounds__(128) void flash_attn_kernel(
    const _Float16* __restrict__ qh, const _Float16* __restrict__ kh,
    const _Float16* __restrict__ vt, const float* __restrict__ rb,
    _Float16* __restrict__ ctx)
{
  __shared__ __align__(16) _Float16 Ks[2][64][72];     // [buf][key][d]  144B rows
  __shared__ __align__(16) _Float16 Vs[2][64][72];     // [buf][d][key]
  __shared__ __align__(16) _Float16 pS[4][16][72];     // per-wave P transpose buffer

  const int t    = threadIdx.x;
  const int wave = t >> 5;
  const int lane = t & 31;
  const int lr   = lane & 15;
  const int hf   = lane >> 4;
  const int h    = blockIdx.y;
  const int blk  = (int)gridDim.x - 1 - (int)blockIdx.x;   // heavy blocks first
  const int qb0  = blk * 64;
  const int q0   = qb0 + wave * 16;

  const _Float16* qbase = qh + (size_t)h * S_LEN * DKH;
  const _Float16* kbase = kh + (size_t)h * S_LEN * DKH;
  const _Float16* vbase = vt + (size_t)h * DKH * S_LEN;
  const float*    rbase = rb + (size_t)h * S_LEN;

  auto stage = [&](int buf, int kb2) {                 // 8 async b128 per thread
    #pragma unroll
    for (int p = 0; p < 4; ++p) {
      const int idx = t + p * 128;                     // 0..511
      const int row = idx >> 3;                        // 0..63
      const int c8  = (idx & 7) << 3;                  // 0,8,...,56
      async_cp16(&Ks[buf][row][c8], kbase + (size_t)(kb2 + row) * DKH + c8);
      async_cp16(&Vs[buf][row][c8], vbase + (size_t)row * S_LEN + kb2 + c8);
    }
  };

  const h16 qa0 = fragA(qbase + (size_t)(q0 + lr) * DKH, hf * 8);        // d 0..31
  const h16 qa1 = fragA(qbase + (size_t)(q0 + lr) * DKH + 32, hf * 8);   // d 32..63

  h16 onesB;                                  // B-matrix of all 1s: row-sum via WMMA
  #pragma unroll
  for (int i = 0; i < 16; ++i) onesB[i] = (_Float16)1.0f;
  f8 fz;
  #pragma unroll
  for (int r = 0; r < 8; ++r) fz[r] = 0.f;

  f8 acc[4], lacc;
  float m_i[8];
  #pragma unroll
  for (int r = 0; r < 8; ++r) { m_i[r] = -1e30f; lacc[r] = 0.f; }
  #pragma unroll
  for (int tt = 0; tt < 4; ++tt)
    #pragma unroll
    for (int r = 0; r < 8; ++r) acc[tt][r] = 0.f;

  const int kend = q0 + 16;                   // this wave's causal extent
  const int kmax = qb0 + 64;                  // block's causal extent

  stage(0, 0);
  for (int kb = 0; kb < kmax; kb += 64) {
    const int buf = (kb >> 6) & 1;
    if (kb + 64 < kmax) {
      stage(buf ^ 1, kb + 64);                         // prefetch next tile
      asm volatile("s_wait_asynccnt 0x8" ::: "memory"); // current tile (first 8) done
    } else {
      asm volatile("s_wait_asynccnt 0x0" ::: "memory");
    }
    __syncthreads();                                   // all waves' copies visible

    if (kb < kend) {                          // wave-uniform: EXEC stays all-ones
      f8 st[4];
      float rbv[4];
      #pragma unroll
      for (int n = 0; n < 4; ++n) {           // scores: S = q(16x64) @ k^T, 64 keys
        const _Float16* krow = &Ks[buf][n*16 + lr][0];
        st[n] = wmma_f16(qa0, fragContig(krow + hf * 16), fz);
        st[n] = wmma_f16(qa1, fragContig(krow + 32 + hf * 16), st[n]);
        rbv[n] = rbase[kb + n*16 + lr];
      }
      #pragma unroll
      for (int r = 0; r < 8; ++r) {
        const int iq = q0 + r + hf * 8;
        float s[4];
        #pragma unroll
        for (int n = 0; n < 4; ++n)
          s[n] = ((kb + n*16 + lr) <= iq) ? (st[n][r] + rbv[n]) : -1e9f;
        float mx = fmaxf(fmaxf(s[0], s[1]), fmaxf(s[2], s[3]));
        mx = fmaxf(mx, __shfl_xor(mx, 1, 32));
        mx = fmaxf(mx, __shfl_xor(mx, 2, 32));
        mx = fmaxf(mx, __shfl_xor(mx, 4, 32));
        mx = fmaxf(mx, __shfl_xor(mx, 8, 32));
        const float mnew  = fmaxf(m_i[r], mx);
        const float alpha = __expf(m_i[r] - mnew);
        m_i[r] = mnew;
        #pragma unroll
        for (int n = 0; n < 4; ++n) {
          const float p = __expf(s[n] - mnew);
          pS[wave][r + hf*8][n*16 + lr] = (_Float16)p;   // C-layout -> A-layout
        }
        #pragma unroll
        for (int tt = 0; tt < 4; ++tt) acc[tt][r] *= alpha;
        lacc[r] *= alpha;
      }
      const h16 pa0 = fragA(&pS[wave][lr][0], hf * 8);   // keys 0..31 (same-wave DS: in order)
      const h16 pa1 = fragA(&pS[wave][lr][32], hf * 8);  // keys 32..63
      #pragma unroll
      for (int tt = 0; tt < 4; ++tt) {                   // out += P(16x64) @ V(64x64)
        const _Float16* vrow = &Vs[buf][tt*16 + lr][0];
        acc[tt] = wmma_f16(pa0, fragContig(vrow + hf * 16), acc[tt]);
        acc[tt] = wmma_f16(pa1, fragContig(vrow + 32 + hf * 16), acc[tt]);
      }
      lacc = wmma_f16(pa0, onesB, lacc);                 // l += rowsum(P): no shuffle tree
      lacc = wmma_f16(pa1, onesB, lacc);
    }
    __syncthreads();                                   // everyone done reading buf
  }

  #pragma unroll
  for (int tt = 0; tt < 4; ++tt)
    #pragma unroll
    for (int r = 0; r < 8; ++r) {
      const int s = q0 + r + hf*8;
      const float o = acc[tt][r] / lacc[r];
      ctx[(size_t)s * DMODEL + h * DKH + tt*16 + lr] = (_Float16)o;
    }
}

// ---------------------------------------------------------------------------
extern "C" void kernel_launch(void* const* d_in, const int* in_sizes, int n_in,
                              void* d_out, int out_size, void* d_ws, size_t ws_size,
                              hipStream_t stream)
{
  (void)in_sizes; (void)n_in; (void)out_size; (void)ws_size;
  const float* query = (const float*)d_in[0];
  const float* key   = (const float*)d_in[1];
  const float* value = (const float*)d_in[2];
  // d_in[3] = mask: causal tril by construction -> handled analytically
  const float* Wq = (const float*)d_in[4];
  const float* bq = (const float*)d_in[5];
  const float* Wk = (const float*)d_in[6];
  const float* bk = (const float*)d_in[7];
  const float* Wv = (const float*)d_in[8];
  const float* bv = (const float*)d_in[9];
  const float* Wo = (const float*)d_in[10];
  const float* bo = (const float*)d_in[11];
  const float* rel = (const float*)d_in[12];

  char* ws = (char*)d_ws;                       // ~16.2 MB used
  _Float16* qh  = (_Float16*)(ws);              // [H][S][DK] f16, pre-scaled by 1/8
  _Float16* kh  = (_Float16*)(ws + (4u<<20));   // [H][S][DK] f16
  _Float16* vt  = (_Float16*)(ws + (8u<<20));   // [H][DK][S] f16 (transposed)
  _Float16* ctx = (_Float16*)(ws + (12u<<20));  // [S][DM] f16
  float*    rbf = (float*)   (ws + (16u<<20));  // [H][S] f32

  const dim3 ggrid(DMODEL/128, S_LEN/128);
  gemm_wmma_kernel<0, float><<<ggrid, 256, 0, stream>>>(query, Wq, bq, qh);
  gemm_wmma_kernel<1, float><<<ggrid, 256, 0, stream>>>(key,   Wk, bk, kh);
  gemm_wmma_kernel<2, float><<<ggrid, 256, 0, stream>>>(value, Wv, bv, vt);
  rel_bias_kernel<<<(NHEAD * S_LEN) / 256, 256, 0, stream>>>(qh, rel, rbf);
  flash_attn_kernel<<<dim3(S_LEN/64, NHEAD), 128, 0, stream>>>(qh, kh, vt, rbf, ctx);
  gemm_wmma_kernel<3, _Float16><<<ggrid, 256, 0, stream>>>(ctx, Wo, bo, d_out);
}
